// HardgroupAttention_11836929868322
// MI455X (gfx1250) — compile-verified, matrix operand order checked by reading, not run on hardware
//
#include <hip/hip_runtime.h>

// ---------------------------------------------------------------------------
// HardgroupAttention for MI455X (gfx1250, wave32, WMMA).
// Fused 3-pass attention: never materializes the 201MB attn tensor.
//   k_cvt      : f32 -> f16 copies of x, W_qkv, W_proj
//   k_qkv      : qkv = x @ W_qkv^T  (WMMA f16, f32 acc); scatters q/k/v [B,h,N,d] f16,
//                plus f32 q_all for the group argmax
//   k_group    : gid[token] = argmax_g (q_all . W_gp[g])   (scalar, precision-critical)
//   k_rowsum   : rowsum[n] = sum_m exp(s[n,m])             (WMMA S recompute)
//   k_colsum   : colsum[m] = sum_n mask*exp(s)/rowsum[n]   (WMMA S recompute)
//   k_attnout  : out = diag(1/rowsum) * (maskexp @ (v/(colsum+eps)))  (2x WMMA)
//   k_proj     : y = out_attn @ W_proj^T                   (WMMA)
// ---------------------------------------------------------------------------

#define NUM_B 4
#define NTOK  1024
#define CDIM  384
#define NH    12
#define HD    32
#define QKVD  1152
#define GPN   20
#define SCALE 0.17677669529663687f  // 32^-0.5

typedef __attribute__((ext_vector_type(16))) _Float16 v16h;
typedef __attribute__((ext_vector_type(8)))  float    v8f;

union U16h { v16h v; unsigned u[8]; _Float16 h[16]; };

// Load a 16x32 (f16) WMMA A/B fragment for one lane.
// k0 points at K=0 of this lane's row/column (K contiguous in memory).
// Per ISA 16-bit layout: lanes 0-15 hold K {0..7, 16..23}, lanes 16-31 hold
// K {8..15, 24..31}; element e of v16h maps to k = (e<8 ? 8*hi+e : 16+8*hi+e-8).
static __device__ inline v16h ld_frag(const _Float16* k0, int hi) {
  const unsigned* p = (const unsigned*)k0;   // paired-f16 32-bit loads
  U16h r;
  const int b2 = hi ? 4 : 0;
#pragma unroll
  for (int j = 0; j < 4; ++j) { r.u[j] = p[b2 + j]; r.u[4 + j] = p[8 + b2 + j]; }
  return r.v;
}

static __device__ inline v8f wmma16(v16h a, v16h b, v8f c) {
  // D = A(16x32) * B(32x16) + C, f32 accumulate
  return __builtin_amdgcn_wmma_f32_16x16x32_f16(false, a, false, b, (short)0, c,
                                                false, false);
}

// ---------------------------------------------------------------------------
__global__ void k_cvt(const float* __restrict__ in, _Float16* __restrict__ out, int n) {
  int i = blockIdx.x * 256 + threadIdx.x;
  if (i < n) out[i] = (_Float16)in[i];
}

// ---------------------------------------------------------------------------
// qkv projection: M=4096 tokens, N=1152, K=384.  One wave per 16x16 tile.
__global__ void k_qkv(const _Float16* __restrict__ xh, const _Float16* __restrict__ wqkvh,
                      _Float16* __restrict__ qh, _Float16* __restrict__ kh,
                      _Float16* __restrict__ vh, float* __restrict__ qf32) {
  const int tid = threadIdx.x, wave = tid >> 5, lane = tid & 31, hi = (lane >> 4) & 1;
  const int mt = blockIdx.x * 8 + wave;      // 0..255 row tiles
  const int ct = blockIdx.y;                 // 0..71  col tiles
  const _Float16* arow = xh    + (size_t)(mt * 16 + (lane & 15)) * CDIM;
  const _Float16* brow = wqkvh + (size_t)(ct * 16 + (lane & 15)) * CDIM;
  v8f acc = {};
#pragma unroll
  for (int ks = 0; ks < CDIM / 32; ++ks) {
    v16h a = ld_frag(arow + ks * 32, hi);
    v16h b = ld_frag(brow + ks * 32, hi);
    acc = wmma16(a, b, acc);
  }
  const int cg = ct * 16 + (lane & 15);
  const int which = cg / CDIM;               // 0=q 1=k 2=v (tile never straddles: 384%16==0)
  const int rem = cg - which * CDIM;
  const int head = rem >> 5, dd = rem & 31;
  _Float16* dst = (which == 0) ? qh : (which == 1) ? kh : vh;
#pragma unroll
  for (int r = 0; r < 8; ++r) {
    const int tok = mt * 16 + hi * 8 + r;    // 0..4095
    const int b = tok >> 10, n = tok & 1023;
    const float val = acc[r];
    dst[((size_t)(b * NH + head) * NTOK + n) * HD + dd] = (_Float16)val;
    if (which == 0) qf32[(size_t)tok * CDIM + rem] = val;
  }
}

// ---------------------------------------------------------------------------
__global__ void k_group(const float* __restrict__ qf32, const float* __restrict__ wgp,
                        int* __restrict__ gid) {
  __shared__ float wsh[GPN * CDIM];
  const int tid = threadIdx.x;
  for (int i = tid; i < GPN * CDIM; i += 256) wsh[i] = wgp[i];
  __syncthreads();
  const int t = blockIdx.x * 256 + tid;      // 0..4095
  const float* q = qf32 + (size_t)t * CDIM;
  float acc[GPN];
#pragma unroll
  for (int g = 0; g < GPN; ++g) acc[g] = 0.f;
  for (int j = 0; j < CDIM; ++j) {
    const float qv = q[j];
#pragma unroll
    for (int g = 0; g < GPN; ++g) acc[g] = fmaf(qv, wsh[g * CDIM + j], acc[g]);
  }
  float best = acc[0]; int bi = 0;
#pragma unroll
  for (int g = 1; g < GPN; ++g) if (acc[g] > best) { best = acc[g]; bi = g; }
  gid[t] = bi;
}

// ---------------------------------------------------------------------------
// rowsum[n] = sum_m exp(scale * q[n].k[m]).  grid (8 rowblks, 12 heads, 4 b).
__global__ void k_rowsum(const _Float16* __restrict__ qh, const _Float16* __restrict__ kh,
                         float* __restrict__ rowsum) {
  const int tid = threadIdx.x, wave = tid >> 5, lane = tid & 31, hi = (lane >> 4) & 1;
  const int head = blockIdx.y, b = blockIdx.z;
  const size_t bh = (size_t)(b * NH + head);
  const int rt = blockIdx.x * 8 + wave;      // row tile 0..63
  const v16h aq = ld_frag(qh + (bh * NTOK + rt * 16 + (lane & 15)) * HD, hi);
  float rs[8] = {};
  for (int mt = 0; mt < 64; ++mt) {
    v16h bk = ld_frag(kh + (bh * NTOK + mt * 16 + (lane & 15)) * HD, hi);
    v8f s = {};
    s = wmma16(aq, bk, s);
#pragma unroll
    for (int r = 0; r < 8; ++r) rs[r] += __expf(s[r] * SCALE);
  }
#pragma unroll
  for (int r = 0; r < 8; ++r) {
#pragma unroll
    for (int m = 1; m < 16; m <<= 1) rs[r] += __shfl_xor(rs[r], m, 32);
  }
  if ((lane & 15) == 0) {
#pragma unroll
    for (int r = 0; r < 8; ++r) rowsum[bh * NTOK + rt * 16 + hi * 8 + r] = rs[r];
  }
}

// ---------------------------------------------------------------------------
// colsum[m] = sum_n mask(n,m) * exp(s[n,m]) / rowsum[n]
__global__ void k_colsum(const _Float16* __restrict__ qh, const _Float16* __restrict__ kh,
                         const float* __restrict__ rowsum, const int* __restrict__ gid,
                         float* __restrict__ colsum) {
  __shared__ float rrec[NTOK];
  __shared__ int gidsh[NTOK];
  const int tid = threadIdx.x, wave = tid >> 5, lane = tid & 31, hi = (lane >> 4) & 1;
  const int head = blockIdx.y, b = blockIdx.z;
  const size_t bh = (size_t)(b * NH + head);
  for (int i = tid; i < NTOK; i += 256) {
    rrec[i] = 1.0f / rowsum[bh * NTOK + i];
    gidsh[i] = gid[b * NTOK + i];
  }
  __syncthreads();
  const int mt = blockIdx.x * 8 + wave;      // col tile 0..63
  const v16h bk = ld_frag(kh + (bh * NTOK + mt * 16 + (lane & 15)) * HD, hi);
  const int gm = gidsh[mt * 16 + (lane & 15)];
  const int same = (head < NH / 2);
  float cs = 0.f;
  for (int nt = 0; nt < 64; ++nt) {
    v16h aq = ld_frag(qh + (bh * NTOK + nt * 16 + (lane & 15)) * HD, hi);
    v8f s = {};
    s = wmma16(aq, bk, s);
#pragma unroll
    for (int r = 0; r < 8; ++r) {
      const int n = nt * 16 + hi * 8 + r;
      const float e = __expf(s[r] * SCALE);
      const bool ok = ((gidsh[n] == gm) == (same != 0));
      cs += ok ? e * rrec[n] : 0.f;
    }
  }
  cs += __shfl_xor(cs, 16, 32);              // combine row halves (same column)
  if (lane < 16) colsum[bh * NTOK + mt * 16 + lane] = cs;
}

// ---------------------------------------------------------------------------
// out_attn[n,:] = (1/rowsum[n]) * sum_m maskexp[n,m] * v[m,:]/(colsum[m]+eps)
__global__ void k_attnout(const _Float16* __restrict__ qh, const _Float16* __restrict__ kh,
                          const _Float16* __restrict__ vh,
                          const float* __restrict__ rowsum, const float* __restrict__ colsum,
                          const int* __restrict__ gid, _Float16* __restrict__ oah) {
  __shared__ __align__(16) _Float16 stage[8][16][32]; // per-wave P tile (C->A relayout)
  __shared__ __align__(16) _Float16 vtT[HD][32];      // v~ transposed: [dd][m_local], K contiguous
  __shared__ float crec[NTOK];
  __shared__ int gidsh[NTOK];
  const int tid = threadIdx.x, wave = tid >> 5, lane = tid & 31, hi = (lane >> 4) & 1;
  const int head = blockIdx.y, b = blockIdx.z;
  const size_t bh = (size_t)(b * NH + head);
  for (int i = tid; i < NTOK; i += 256) {
    crec[i] = 1.0f / (colsum[bh * NTOK + i] + 1e-8f);
    gidsh[i] = gid[b * NTOK + i];
  }
  __syncthreads();
  const int rt = blockIdx.x * 8 + wave;      // row tile 0..63
  const v16h aq = ld_frag(qh + (bh * NTOK + rt * 16 + (lane & 15)) * HD, hi);
  const int same = (head < NH / 2);
  v8f o0 = {}, o1 = {};
  for (int mc = 0; mc < 32; ++mc) {          // 32 columns of K at a time
    const int m0 = mc * 32;
    for (int i = tid; i < HD * 32; i += 256) {          // build v~^T tile
      const int dd = i >> 5, ml = i & 31;
      vtT[dd][ml] = (_Float16)((float)vh[(bh * NTOK + m0 + ml) * HD + dd] * crec[m0 + ml]);
    }
    __syncthreads();
#pragma unroll
    for (int t = 0; t < 2; ++t) {            // two 16-col S tiles -> stage (masked exp)
      const int mtile = m0 + t * 16;
      v16h bk = ld_frag(kh + (bh * NTOK + mtile + (lane & 15)) * HD, hi);
      v8f s = {};
      s = wmma16(aq, bk, s);
      const int gm = gidsh[mtile + (lane & 15)];
#pragma unroll
      for (int r = 0; r < 8; ++r) {
        const int n = rt * 16 + hi * 8 + r;
        const float e = __expf(s[r] * SCALE);
        const bool ok = ((gidsh[n] == gm) == (same != 0));
        stage[wave][hi * 8 + r][t * 16 + (lane & 15)] = (_Float16)(ok ? e : 0.f);
      }
    }
    __syncthreads();
    v16h ap  = ld_frag(&stage[wave][lane & 15][0], hi);  // P tile, A layout (K=m)
    v16h bv0 = ld_frag(&vtT[(lane & 15)][0], hi);        // dd 0..15
    v16h bv1 = ld_frag(&vtT[16 + (lane & 15)][0], hi);   // dd 16..31
    o0 = wmma16(ap, bv0, o0);
    o1 = wmma16(ap, bv1, o1);
    __syncthreads();                          // protect vtT before next build
  }
#pragma unroll
  for (int r = 0; r < 8; ++r) {
    const int n = rt * 16 + hi * 8 + r;
    const float rr = 1.0f / rowsum[bh * NTOK + n];
    const size_t tok = (size_t)(b * NTOK + n);
    oah[tok * CDIM + head * HD + (lane & 15)]      = (_Float16)(o0[r] * rr);
    oah[tok * CDIM + head * HD + 16 + (lane & 15)] = (_Float16)(o1[r] * rr);
  }
}

// ---------------------------------------------------------------------------
// y = out_attn(4096x384) @ W_proj^T(384x384), f32 out
__global__ void k_proj(const _Float16* __restrict__ oah, const _Float16* __restrict__ wprojh,
                       float* __restrict__ out) {
  const int tid = threadIdx.x, wave = tid >> 5, lane = tid & 31, hi = (lane >> 4) & 1;
  const int mt = blockIdx.x * 8 + wave;      // 0..255
  const int ct = blockIdx.y;                 // 0..23
  const _Float16* arow = oah    + (size_t)(mt * 16 + (lane & 15)) * CDIM;
  const _Float16* brow = wprojh + (size_t)(ct * 16 + (lane & 15)) * CDIM;
  v8f acc = {};
#pragma unroll
  for (int ks = 0; ks < CDIM / 32; ++ks) {
    v16h a = ld_frag(arow + ks * 32, hi);
    v16h b = ld_frag(brow + ks * 32, hi);
    acc = wmma16(a, b, acc);
  }
#pragma unroll
  for (int r = 0; r < 8; ++r)
    out[(size_t)(mt * 16 + hi * 8 + r) * CDIM + ct * 16 + (lane & 15)] = acc[r];
}

// ---------------------------------------------------------------------------
extern "C" void kernel_launch(void* const* d_in, const int* in_sizes, int n_in,
                              void* d_out, int out_size, void* d_ws, size_t ws_size,
                              hipStream_t stream) {
  const float* x      = (const float*)d_in[0];
  const float* W_qkv  = (const float*)d_in[1];
  const float* W_proj = (const float*)d_in[2];
  const float* W_gp   = (const float*)d_in[3];
  float* out = (float*)d_out;
  char* ws = (char*)d_ws;

  constexpr size_t TOKS = (size_t)NUM_B * NTOK;          // 4096
  constexpr size_t OFF_XH   = 0;                                        // f16 x
  constexpr size_t OFF_WQKV = OFF_XH   + TOKS * CDIM * 2;               // f16 W_qkv
  constexpr size_t OFF_WPRJ = OFF_WQKV + (size_t)QKVD * CDIM * 2;       // f16 W_proj
  constexpr size_t OFF_QH   = OFF_WPRJ + (size_t)CDIM * CDIM * 2;       // f16 q [B,h,N,d]
  constexpr size_t OFF_KH   = OFF_QH + (size_t)NUM_B * NH * NTOK * HD * 2;
  constexpr size_t OFF_VH   = OFF_KH + (size_t)NUM_B * NH * NTOK * HD * 2;
  constexpr size_t OFF_QF   = OFF_VH + (size_t)NUM_B * NH * NTOK * HD * 2; // f32 q_all
  constexpr size_t OFF_GID  = OFF_QF + TOKS * CDIM * 4;
  constexpr size_t OFF_RS   = OFF_GID + TOKS * 4;
  constexpr size_t OFF_CS   = OFF_RS + (size_t)NUM_B * NH * NTOK * 4;
  constexpr size_t OFF_OA   = OFF_CS + (size_t)NUM_B * NH * NTOK * 4;   // f16 out_attn

  _Float16* xh     = (_Float16*)(ws + OFF_XH);
  _Float16* wqkvh  = (_Float16*)(ws + OFF_WQKV);
  _Float16* wprojh = (_Float16*)(ws + OFF_WPRJ);
  _Float16* qh     = (_Float16*)(ws + OFF_QH);
  _Float16* kh     = (_Float16*)(ws + OFF_KH);
  _Float16* vh     = (_Float16*)(ws + OFF_VH);
  float*    qf32   = (float*)(ws + OFF_QF);
  int*      gid    = (int*)(ws + OFF_GID);
  float*    rowsum = (float*)(ws + OFF_RS);
  float*    colsum = (float*)(ws + OFF_CS);
  _Float16* oah    = (_Float16*)(ws + OFF_OA);

  const int nx = (int)(TOKS * CDIM), nq = QKVD * CDIM, np = CDIM * CDIM;
  k_cvt<<<(nx + 255) / 256, 256, 0, stream>>>(x, xh, nx);
  k_cvt<<<(nq + 255) / 256, 256, 0, stream>>>(W_qkv, wqkvh, nq);
  k_cvt<<<(np + 255) / 256, 256, 0, stream>>>(W_proj, wprojh, np);

  k_qkv<<<dim3(32, QKVD / 16), 256, 0, stream>>>(xh, wqkvh, qh, kh, vh, qf32);
  k_group<<<dim3(16), 256, 0, stream>>>(qf32, W_gp, gid);
  k_rowsum<<<dim3(8, NH, NUM_B), 256, 0, stream>>>(qh, kh, rowsum);
  k_colsum<<<dim3(8, NH, NUM_B), 256, 0, stream>>>(qh, kh, rowsum, gid, colsum);
  k_attnout<<<dim3(8, NH, NUM_B), 256, 0, stream>>>(qh, kh, vh, rowsum, colsum, gid, oah);
  k_proj<<<dim3(32, CDIM / 16), 256, 0, stream>>>(oah, wprojh, out);
}
